// QLinear_27015344292266
// MI455X (gfx1250) — matrix operand order repaired and checked by reading, not compiled, not run on hardware
//
#include <hip/hip_runtime.h>

typedef __attribute__((ext_vector_type(16))) __bf16 v16bf;
typedef __attribute__((ext_vector_type(8)))  __bf16 v8bf;
typedef __attribute__((ext_vector_type(4)))  __bf16 v4bf;
typedef __attribute__((ext_vector_type(8)))  float  v8f;

#define N_DIM   1024     // inner dim K and output dim N
#define M_DIM   32768    // batch rows
#define Q_PARAM 4.0f
#define SCALE   1.2f     // BETA(0.3) * D_LAT(4)

// ---------------------------------------------------------------------------
// D4 lattice nearest-point quantizer (Conway–Sloane): round each coord; if
// the rounded sum is odd, push the worst coordinate to its 2nd-nearest int.
// ---------------------------------------------------------------------------
__device__ __forceinline__ void q_d4(const float x[4], float out[4]) {
    float f[4], delta[4];
    float s = 0.f;
#pragma unroll
    for (int i = 0; i < 4; ++i) {
        f[i] = rintf(x[i]);          // round-to-nearest-even, matches jnp.round
        delta[i] = x[i] - f[i];
        s += f[i];
    }
    int idx = 0;
    float best = fabsf(delta[0]);
#pragma unroll
    for (int i = 1; i < 4; ++i) {
        float a = fabsf(delta[i]);
        if (a > best) { best = a; idx = i; }
    }
    const bool odd = (((long long)llrintf(s)) & 1LL) != 0;
#pragma unroll
    for (int i = 0; i < 4; ++i) {
        float g = f[i];
        if (i == idx) g += (delta[i] >= 0.f ? 1.f : -1.f);
        out[i] = odd ? g : f[i];
    }
}

// ---------------------------------------------------------------------------
// Kernel 1: per-row norm-scaled hierarchical nested-lattice quantize/dequant
// of the weight matrix, emitting bf16. One block per row (1024 rows); each of
// the 256 threads owns one D4 group of 4 columns.
// ---------------------------------------------------------------------------
__global__ void quantize_weight_kernel(const float* __restrict__ w,
                                       __bf16* __restrict__ wq) {
    __shared__ float red[256];
    const int row = blockIdx.x;
    const int t   = threadIdx.x;

    const float4 v = ((const float4*)(w + (size_t)row * N_DIM))[t];
    red[t] = v.x * v.x + v.y * v.y + v.z * v.z + v.w * v.w;
    __syncthreads();
#pragma unroll
    for (int off = 128; off > 0; off >>= 1) {
        if (t < off) red[t] += red[t + off];
        __syncthreads();
    }
    const float norm_safe = fmaxf(sqrtf(red[0]), 1e-8f);
    const float inv = SCALE / norm_safe;

    float xt[4]   = { v.x * inv, v.y * inv, v.z * inv, v.w * inv };
    float xhat[4] = { 0.f, 0.f, 0.f, 0.f };

    float pw = 1.f;
#pragma unroll
    for (int m = 0; m < 2; ++m) {                  // M_LEVELS = 2
        float qx[4];  q_d4(xt, qx);
        float sc[4];
#pragma unroll
        for (int i = 0; i < 4; ++i) sc[i] = qx[i] * (1.f / Q_PARAM);
        float qd[4];  q_d4(sc, qd);
#pragma unroll
        for (int i = 0; i < 4; ++i) {
            xhat[i] += pw * (qx[i] - Q_PARAM * qd[i]);   // coset representative
            xt[i] = sc[i];
        }
        pw *= Q_PARAM;
    }

    const float rescale = norm_safe / SCALE;
    v4bf o;
#pragma unroll
    for (int i = 0; i < 4; ++i) o[i] = (__bf16)(xhat[i] * rescale);
    *(v4bf*)(wq + (size_t)row * N_DIM + 4 * t) = o;
}

// ---------------------------------------------------------------------------
// Kernel 2: x f32 -> bf16. f32 x is single-use -> non-temporal loads; the
// bf16 result (64 MB) is left with default (RT) hint so it stays in the
// 192 MB L2 for the GEMM.
// ---------------------------------------------------------------------------
__global__ void convert_x_kernel(const float* __restrict__ x,
                                 __bf16* __restrict__ xb) {
    const size_t i = ((size_t)blockIdx.x * blockDim.x + threadIdx.x) * 8;
    v8bf o;
#pragma unroll
    for (int j = 0; j < 8; ++j)
        o[j] = (__bf16)__builtin_nontemporal_load(x + i + j);
    *(v8bf*)(xb + i) = o;
}

// ---------------------------------------------------------------------------
// Kernel 3: bf16 WMMA GEMM.  C[m,n] = sum_k A[m,k] * W[n,k] + bias[n].
// 256 threads = 8 waves arranged 2(M) x 4(N); wave tile 64x64 via 4x4
// v_wmma_f32_16x16x32_bf16 accumulators; K-loop of 32 steps.
// Per K-step: 8 fragments (16 b128 loads) feed 16 WMMAs -> 1 load/wmma.
//
// Fragment layouts as loaded here (wave32; contiguous k, not the ISA table order):
//   A 16x32 bf16: lane l holds row M=l%16, K = (l/16)*16 .. +16  (contiguous, same order as B)
//   B 32x16 bf16: lane l holds col N=l%16, K = (l/16)*16 .. +16  (contiguous,
//                 which is exactly a row-major run of W[n][k])
// All fragment addresses fold into 24-bit immediate offsets off 2 base ptrs.
// ---------------------------------------------------------------------------
__global__ void __launch_bounds__(256)
gemm_bf16_wmma(const __bf16* __restrict__ A,    // [M_DIM][N_DIM] bf16
               const __bf16* __restrict__ B,    // [N_DIM][N_DIM] bf16 (quantized W)
               const float* __restrict__ bias,  // [N_DIM]
               float* __restrict__ C) {         // [M_DIM][N_DIM] f32
    const int lane = threadIdx.x & 31;
    const int wave = threadIdx.x >> 5;
    const int l16  = lane & 15;
    const int hs   = lane >> 4;                       // half-select (0/1)

    const int mBase = blockIdx.y * 128 + (wave >> 2) * 64;   // 2 waves in M
    const int nBase = blockIdx.x * 256 + (wave & 3) * 64;    // 4 waves in N

    v8f acc[4][4] = {};

    const __bf16* aP = A + (size_t)(mBase + l16) * N_DIM + hs * 16;
    const __bf16* bP = B + (size_t)(nBase + l16) * N_DIM + hs * 16;

#pragma unroll 2
    for (int kb = 0; kb < N_DIM / 32; ++kb) {
        const int ko = kb * 32;

        v16bf a[4], b[4];
#pragma unroll
        for (int mt = 0; mt < 4; ++mt) {
            const __bf16* p = aP + (size_t)mt * 16 * N_DIM + ko;
            const v8bf lo = *(const v8bf*)(p);
            const v8bf hi = *(const v8bf*)(p + 8);
            a[mt] = __builtin_shufflevector(lo, hi,
                0,1,2,3,4,5,6,7,8,9,10,11,12,13,14,15);
        }
#pragma unroll
        for (int nt = 0; nt < 4; ++nt)
            b[nt] = *(const v16bf*)(bP + (size_t)nt * 16 * N_DIM + ko);

#pragma unroll
        for (int mt = 0; mt < 4; ++mt)
#pragma unroll
            for (int nt = 0; nt < 4; ++nt)
                acc[mt][nt] = __builtin_amdgcn_wmma_f32_16x16x32_bf16(
                    false, a[mt], false, b[nt], (short)0, acc[mt][nt],
                    false, false);
    }

    // Epilogue: D layout — lane group g=lane/16: VGPR r holds row M = g*8 + r,
    // col N = lane%16. 16 consecutive lanes write 64 contiguous bytes.
    // Output is never re-read -> non-temporal stores keep xb/wq in L2.
#pragma unroll
    for (int nt = 0; nt < 4; ++nt) {
        const int col = nBase + nt * 16 + l16;
        const float bv = bias[col];
#pragma unroll
        for (int mt = 0; mt < 4; ++mt) {
#pragma unroll
            for (int r = 0; r < 8; ++r) {
                const int row = mBase + mt * 16 + hs * 8 + r;
                __builtin_nontemporal_store(acc[mt][nt][r] + bv,
                                            &C[(size_t)row * N_DIM + col]);
            }
        }
    }
}

// ---------------------------------------------------------------------------
extern "C" void kernel_launch(void* const* d_in, const int* in_sizes, int n_in,
                              void* d_out, int out_size, void* d_ws, size_t ws_size,
                              hipStream_t stream) {
    const float* x    = (const float*)d_in[0];   // [32768,1024] f32
    const float* w    = (const float*)d_in[1];   // [1024,1024]  f32
    const float* bias = (const float*)d_in[2];   // [1024]       f32
    float* out = (float*)d_out;                  // [32768,1024] f32

    __bf16* wq = (__bf16*)d_ws;                                        // 2 MB
    __bf16* xb = (__bf16*)((char*)d_ws + (size_t)N_DIM * N_DIM * 2);   // 64 MB

    // 1) HNLQ quantize/dequant weight -> bf16
    quantize_weight_kernel<<<N_DIM, 256, 0, stream>>>(w, wq);

    // 2) x f32 -> bf16 (8 elements/thread)
    const int cvt_blocks = (int)(((size_t)M_DIM * N_DIM / 8) / 256);   // 16384
    convert_x_kernel<<<cvt_blocks, 256, 0, stream>>>(x, xb);

    // 3) WMMA GEMM: grid = (N/256, M/128), 128x256 block tile, 64x64/wave
    gemm_bf16_wmma<<<dim3(N_DIM / 256, M_DIM / 128), 256, 0, stream>>>(
        xb, wq, bias, out);
}
